// MultiHeadAttention_40664750359017
// MI455X (gfx1250) — matrix operand order repaired and checked by reading, not compiled, not run on hardware
//
#include <hip/hip_runtime.h>

// ---------------------------------------------------------------------------
// MI455X / gfx1250 causal multi-head attention
//   B=2, T=2048, C=1024, H=16, HD=64
//   bf16 WMMA (v_wmma_f32_16x16x32_bf16), f32 accumulate.
//   TDM (tensor_load_to_lds) stages row-major tiles with HW padding;
//   transposed B/V tiles staged through registers; double-buffered LDS.
// ---------------------------------------------------------------------------

typedef __attribute__((ext_vector_type(16))) __bf16 v16bf;
typedef __attribute__((ext_vector_type(8)))  float  v8f;
typedef __attribute__((ext_vector_type(4)))  unsigned int v4u;
typedef __attribute__((ext_vector_type(8)))  int    v8i_t;
typedef __attribute__((ext_vector_type(4)))  int    v4i_t;

typedef unsigned short u16;
typedef unsigned int   u32;
typedef unsigned long long u64;

#define Bsz 2
#define Tsz 2048
#define Csz 1024
#define Hn  16
#define HD  64

__device__ __forceinline__ u16 f2bf(float f) {
  u32 x = __float_as_uint(f);
  x += 0x7FFFu + ((x >> 16) & 1u);       // round-to-nearest-even
  return (u16)(x >> 16);
}

union AFrag {
  v16bf v;
  u16   s[16];
  u32   u[8];
};

__device__ __forceinline__ v8f wmma_bf16(const AFrag& a, const AFrag& b, v8f c) {
  return __builtin_amdgcn_wmma_f32_16x16x32_bf16(
      /*neg_a=*/false, a.v, /*neg_b=*/false, b.v,
      /*c_mod=*/(short)0, c, /*reuse_a=*/false, /*reuse_b=*/false);
}

__device__ __forceinline__ v8f v8f_zero() {
  return (v8f){0.f, 0.f, 0.f, 0.f, 0.f, 0.f, 0.f, 0.f};
}

// ---------------------------------------------------------------------------
// Tensor Data Mover: 2D tile (tile_d1 rows x tile_d0 elems, 2B elems) from
// row-major global (stride0 elems) into LDS with 1-DWORD row padding.
// D# layout per CDNA5 ISA §8.3/8.4 (group0 128b, group1 256b).
// ---------------------------------------------------------------------------
union D128u { v4u  v; u64 q[2]; };
union D256i { v8i_t v; u64 q[4]; };
union D128i { v4i_t v; u64 q[2]; };

__device__ __forceinline__ void tdm_load_2d(const void* gptr, u32 lds_off,
                                            u32 tile_d0, u32 tile_d1,
                                            u32 tensor_d0, u32 tensor_d1,
                                            u32 stride0, u32 pad_interval_code) {
  u64 ga = (u64)(uintptr_t)gptr;
  D128u g0; D256i g1; D128i g2, g3;
  // group0: count=1 | lds_addr[63:32] ; global_addr[120:64] | type=2 [127:126]
  g0.q[0] = 1ull | ((u64)lds_off << 32);
  g0.q[1] = (ga & ((1ull << 57) - 1)) | (2ull << 62);
  // group1: data_size=1(2B)[17:16], pad_enable[20], pad_interval[24:22],
  //         pad_amount=0(1 DWORD)[31:25], tensor_dim0[79:48],
  //         tensor_dim1[111:80], tile_dim0[127:112], tile_dim1[143:128],
  //         tensor_dim0_stride[207:160]
  g1.q[0] = (1ull << 16) | (1ull << 20) | ((u64)pad_interval_code << 22) |
            ((u64)(tensor_d0 & 0xFFFFu) << 48);
  g1.q[1] = ((u64)tensor_d0 >> 16) | ((u64)tensor_d1 << 16) | ((u64)tile_d0 << 48);
  g1.q[2] = (u64)tile_d1 | ((u64)stride0 << 32);
  g1.q[3] = 0;
  g2.q[0] = g2.q[1] = 0;
  g3.q[0] = g3.q[1] = 0;
#if __clang_major__ >= 23
  D256i gz; gz.q[0] = gz.q[1] = gz.q[2] = gz.q[3] = 0;
  __builtin_amdgcn_tensor_load_to_lds(g0.v, g1.v, g2.v, g3.v, gz.v, 0);
#else
  __builtin_amdgcn_tensor_load_to_lds(g0.v, g1.v, g2.v, g3.v, 0);
#endif
}

// ---------------------------------------------------------------------------
// Kernel 0: f32 -> bf16, 4 elements/thread
// ---------------------------------------------------------------------------
__global__ __launch_bounds__(256) void cvt_bf16x4(const float4* __restrict__ src,
                                                  uint2* __restrict__ dst, int n4) {
  int i = blockIdx.x * blockDim.x + threadIdx.x;
  if (i < n4) {
    float4 f = src[i];
    uint2 o;
    o.x = (u32)f2bf(f.x) | ((u32)f2bf(f.y) << 16);
    o.y = (u32)f2bf(f.z) | ((u32)f2bf(f.w) << 16);
    dst[i] = o;
  }
}

// ---------------------------------------------------------------------------
// GEMM: C[M,N] = A[M,K] bf16 x B[K,N] bf16 + bias
//   128 threads = 4 waves; workgroup tile 64x128; wave tile 32x64 (2x4 WMMA).
//   A tile staged by TDM (HW row-pad -> stride 34); B staged transposed
//   through registers. Double-buffered LDS, one barrier per K-step.
//   MODE 0: qkv epilogue -> scatter bf16 q(scaled)/k/v into [B,H,T,HD]
//   MODE 1: proj epilogue -> f32 output [M,N]
// ---------------------------------------------------------------------------
template <int MODE>
__global__ __launch_bounds__(128) void gemm_bf16_kernel(
    const u16* __restrict__ A, const u16* __restrict__ Bw,
    const float* __restrict__ bias,
    u16* __restrict__ qO, u16* __restrict__ kO, u16* __restrict__ vO,
    float* __restrict__ fO,
    int M, int N, int K) {
  __shared__ u16 As[2][64][34];    // [m][k], TDM-padded rows (16 DW + 1 DW)
  __shared__ u16 Bt[2][128][34];   // transposed: [n][k]

  const int tid  = threadIdx.x;
  const int lane = tid & 31;
  const int wv   = tid >> 5;
  const int wr   = wv >> 1, wc = wv & 1;
  const int wgM  = blockIdx.y * 64, wgN = blockIdx.x * 128;
  const int hl   = lane >> 4;
  const int l16  = lane & 15;

  const int bKr = (tid >> 4) * 2, bN0 = (tid & 15) * 8;   // B staging geometry

  auto tdmA = [&](int k0, int buf) {
    if (wv == 0)
      tdm_load_2d(A + (size_t)wgM * K + k0,
                  (u32)(uintptr_t)&As[buf][0][0],
                  /*tile_d0=*/32, /*tile_d1=*/64,
                  /*tensor_d0=*/(u32)K, /*tensor_d1=*/(u32)M,
                  /*stride0=*/(u32)K, /*pad_interval(16 DW)=*/3);
  };
  auto loadB = [&](int k0, uint4* rB) {
#pragma unroll
    for (int cc = 0; cc < 2; ++cc) {
      const u16* p = Bw + (size_t)(k0 + bKr + cc * 16) * N + wgN + bN0;
      rB[2 * cc]     = *(const uint4*)p;
      rB[2 * cc + 1] = *(const uint4*)(p + N);
    }
  };
  auto storeB = [&](int buf, const uint4* rB) {
#pragma unroll
    for (int cc = 0; cc < 2; ++cc) {
      int kr = bKr + cc * 16;
      u32 lo[4] = {rB[2 * cc].x, rB[2 * cc].y, rB[2 * cc].z, rB[2 * cc].w};
      u32 hi[4] = {rB[2 * cc + 1].x, rB[2 * cc + 1].y, rB[2 * cc + 1].z, rB[2 * cc + 1].w};
#pragma unroll
      for (int i2 = 0; i2 < 4; ++i2) {
        u32 a = lo[i2], b = hi[i2];
        *(u32*)&Bt[buf][bN0 + 2 * i2][kr]     = (a & 0xffffu) | (b << 16);
        *(u32*)&Bt[buf][bN0 + 2 * i2 + 1][kr] = (a >> 16) | (b & 0xffff0000u);
      }
    }
  };

  v8f acc[2][4];
#pragma unroll
  for (int i = 0; i < 2; ++i)
#pragma unroll
    for (int j = 0; j < 4; ++j) acc[i][j] = v8f_zero();

  uint4 rB[4];
  tdmA(0, 0);
  loadB(0, rB);
  storeB(0, rB);
  if (wv == 0) __builtin_amdgcn_s_wait_tensorcnt(0);
  int cur = 0;
  const int kSteps = K / 32;

  for (int ks = 0; ks < kSteps; ++ks) {
    __syncthreads();
    const bool hasNext = (ks + 1) < kSteps;
    if (hasNext) {
      tdmA((ks + 1) * 32, cur ^ 1);          // DMA overlaps the WMMAs below
      loadB((ks + 1) * 32, rB);
      if ((ks + 2) < kSteps)                 // deep prefetch -> global_prefetch_b8
        __builtin_prefetch(Bw + (size_t)((ks + 2) * 32 + bKr) * N + wgN + bN0, 0, 1);
    }

    // A fragments: lane half 0 -> K 0..15 ; half 1 -> K 16..31
    AFrag a[2];
#pragma unroll
    for (int i = 0; i < 2; ++i) {
      int m = wr * 32 + i * 16 + l16;
      int ka = hl * 16;
#pragma unroll
      for (int j = 0; j < 4; ++j) {
        a[i].u[j]     = *(const u32*)&As[cur][m][ka + 2 * j];
        a[i].u[j + 4] = *(const u32*)&As[cur][m][ka + 8 + 2 * j];
      }
    }
    // B fragments from transposed tile: lane = N, b32 K-pairs
#pragma unroll
    for (int j = 0; j < 4; ++j) {
      AFrag b;
      int n = wc * 64 + j * 16 + l16;
      int kb = hl * 16;
#pragma unroll
      for (int pp = 0; pp < 8; ++pp)
        b.u[pp] = *(const u32*)&Bt[cur][n][kb + 2 * pp];
#pragma unroll
      for (int i = 0; i < 2; ++i) acc[i][j] = wmma_bf16(a[i], b, acc[i][j]);
    }

    if (hasNext) {
      storeB(cur ^ 1, rB);
      if (wv == 0) __builtin_amdgcn_s_wait_tensorcnt(0);
      cur ^= 1;
    }
  }

  // Epilogue. C/D layout: N = l16, M = hl*8 + r  (per 16x16 tile)
#pragma unroll
  for (int i = 0; i < 2; ++i) {
#pragma unroll
    for (int j = 0; j < 4; ++j) {
      int nG = wgN + wc * 64 + j * 16 + l16;
      float bv = bias[nG];
#pragma unroll
      for (int r = 0; r < 8; ++r) {
        int mG = wgM + wr * 32 + i * 16 + hl * 8 + r;
        float val = acc[i][j][r] + bv;
        if (MODE == 0) {
          int h = nG / (3 * HD), rr = nG % (3 * HD);
          int bb = mG >> 11, t = mG & (Tsz - 1);
          size_t base = ((size_t)(bb * Hn + h) * Tsz + t) * HD;
          if (rr < HD)          qO[base + rr]          = f2bf(val * 0.125f);
          else if (rr < 2 * HD) kO[base + rr - HD]     = f2bf(val);
          else                  vO[base + rr - 2 * HD] = f2bf(val);
        } else {
          fO[(size_t)mG * N + nG] = val;
        }
      }
    }
  }
}

// ---------------------------------------------------------------------------
// Flash attention: grid = B*H*(T/64) blocks, 4 waves.
//   Wave: 16 q-rows x 64 dims accumulator, online softmax, 32-key steps.
//   K tile via TDM (row-pad 32 DW + 1 -> stride 66); V tile transposed [d][k']
//   through registers. Double-buffered LDS, one barrier per step.
// ---------------------------------------------------------------------------
__global__ __launch_bounds__(128) void attn_kernel(const u16* __restrict__ Q,
                                                   const u16* __restrict__ Kt,
                                                   const u16* __restrict__ Vin,
                                                   u16* __restrict__ O) {
  __shared__ u16 Ks[2][32][66];
  __shared__ u16 Vt[2][64][34];    // transposed: [d][k']
  __shared__ u16 Ps[4][16][34];    // per-wave P relayout buffer

  const int tid  = threadIdx.x;
  const int lane = tid & 31;
  const int wv   = tid >> 5;
  const int hl   = lane >> 4, l16 = lane & 15;

  const int qt = blockIdx.x & 31;          // q tile (T/64 = 32)
  const int bh = blockIdx.x >> 5;
  const int bb = bh >> 4, hh = bh & 15;

  const u16* qp = Q   + (size_t)bh * Tsz * HD;
  const u16* kp = Kt  + (size_t)bh * Tsz * HD;
  const u16* vp = Vin + (size_t)bh * Tsz * HD;

  const int qBase = qt * 64 + wv * 16;
  const int vKr = (tid >> 3) * 2, vN0 = (tid & 7) * 8;   // V staging geometry

  auto tdmK = [&](int kOff, int buf) {
    if (wv == 0)
      tdm_load_2d(kp + (size_t)kOff * HD,
                  (u32)(uintptr_t)&Ks[buf][0][0],
                  /*tile_d0=*/64, /*tile_d1=*/32,
                  /*tensor_d0=*/64, /*tensor_d1=*/Tsz,
                  /*stride0=*/64, /*pad_interval(32 DW)=*/4);
  };
  auto loadV = [&](int kOff, uint4* rV) {
    const u16* p = vp + (size_t)(kOff + vKr) * HD + vN0;
    rV[0] = *(const uint4*)p;
    rV[1] = *(const uint4*)(p + HD);
  };
  auto storeV = [&](int buf, const uint4* rV) {
    u32 lo[4] = {rV[0].x, rV[0].y, rV[0].z, rV[0].w};
    u32 hi[4] = {rV[1].x, rV[1].y, rV[1].z, rV[1].w};
#pragma unroll
    for (int i2 = 0; i2 < 4; ++i2) {
      u32 a = lo[i2], b = hi[i2];
      *(u32*)&Vt[buf][vN0 + 2 * i2][vKr]     = (a & 0xffffu) | (b << 16);
      *(u32*)&Vt[buf][vN0 + 2 * i2 + 1][vKr] = (a >> 16) | (b & 0xffff0000u);
    }
  };

  // Preload Q fragments for both 32-wide d halves (b32 global loads)
  AFrag aQ[2];
  {
    int row = qBase + l16;
#pragma unroll
    for (int dh = 0; dh < 2; ++dh) {
      int ka = dh * 32 + hl * 16;
#pragma unroll
      for (int j = 0; j < 4; ++j) {
        aQ[dh].u[j]     = *(const u32*)(qp + (size_t)row * HD + ka + 2 * j);
        aQ[dh].u[j + 4] = *(const u32*)(qp + (size_t)row * HD + ka + 8 + 2 * j);
      }
    }
  }

  v8f accO[4];
#pragma unroll
  for (int dt = 0; dt < 4; ++dt) accO[dt] = v8f_zero();
  float mrow[8], lrow[8];
#pragma unroll
  for (int r = 0; r < 8; ++r) { mrow[r] = -1.0e30f; lrow[r] = 0.f; }

  const int nsteps = (qt + 1) * 2;         // 32 keys per step, causal bound
  uint4 rV[2];
  tdmK(0, 0);
  loadV(0, rV);
  storeV(0, rV);
  if (wv == 0) __builtin_amdgcn_s_wait_tensorcnt(0);
  int cur = 0;

  for (int kk = 0; kk < nsteps; ++kk) {
    const int kOff = kk * 32;
    __syncthreads();
    const bool hasNext = (kk + 1) < nsteps;
    if (hasNext) {
      tdmK((kk + 1) * 32, cur ^ 1);
      loadV((kk + 1) * 32, rV);
    }

    // S = Q K^T : 2 tiles of 16 keys; B frag lane = key, b32 d-pairs
    float s0[8], s1[8];
    const bool needMask = (kOff + 31) > qBase + 15;  // wave-uniform
#pragma unroll
    for (int sj = 0; sj < 2; ++sj) {
      v8f accS = v8f_zero();
#pragma unroll
      for (int dh = 0; dh < 2; ++dh) {
        AFrag bK;
        int krow = sj * 16 + l16;
        int kb   = dh * 32 + hl * 16;
#pragma unroll
        for (int pp = 0; pp < 8; ++pp)
          bK.u[pp] = *(const u32*)&Ks[cur][krow][kb + 2 * pp];
        accS = wmma_bf16(aQ[dh], bK, accS);
      }
      float* sd = sj ? s1 : s0;
      if (needMask) {
        int kcol = kOff + sj * 16 + l16;
#pragma unroll
        for (int r = 0; r < 8; ++r) {
          int qrow = qBase + hl * 8 + r;
          sd[r] = (kcol <= qrow) ? accS[r] : -3.0e38f;   // causal mask
        }
      } else {
#pragma unroll
        for (int r = 0; r < 8; ++r) sd[r] = accS[r];
      }
    }

    // Online softmax per row (row's 16 cols live across a 16-lane half)
#pragma unroll
    for (int r = 0; r < 8; ++r) {
      float mt = fmaxf(s0[r], s1[r]);
#pragma unroll
      for (int off = 1; off < 16; off <<= 1)
        mt = fmaxf(mt, __shfl_xor(mt, off, 32));
      float mnew  = fmaxf(mrow[r], mt);
      float scale = __expf(mrow[r] - mnew);
      float p0 = __expf(s0[r] - mnew);
      float p1 = __expf(s1[r] - mnew);
      float ps = p0 + p1;
#pragma unroll
      for (int off = 1; off < 16; off <<= 1)
        ps += __shfl_xor(ps, off, 32);
      lrow[r] = lrow[r] * scale + ps;
      mrow[r] = mnew;
#pragma unroll
      for (int dt = 0; dt < 4; ++dt) accO[dt][r] *= scale;
      int prow = hl * 8 + r;
      Ps[wv][prow][l16]      = f2bf(p0);
      Ps[wv][prow][16 + l16] = f2bf(p1);
    }

    // O += P @ V  (A frag of P via LDS relayout; B frags from transposed Vt)
    AFrag aP;
    {
      int ka = hl * 16;
#pragma unroll
      for (int j = 0; j < 4; ++j) {
        aP.u[j]     = *(const u32*)&Ps[wv][l16][ka + 2 * j];
        aP.u[j + 4] = *(const u32*)&Ps[wv][l16][ka + 8 + 2 * j];
      }
    }
#pragma unroll
    for (int dt = 0; dt < 4; ++dt) {
      AFrag bV;
      int n  = dt * 16 + l16;
      int kb = hl * 16;
#pragma unroll
      for (int pp = 0; pp < 8; ++pp)
        bV.u[pp] = *(const u32*)&Vt[cur][n][kb + 2 * pp];
      accO[dt] = wmma_bf16(aP, bV, accO[dt]);
    }

    if (hasNext) {
      storeV(cur ^ 1, rV);
      if (wv == 0) __builtin_amdgcn_s_wait_tensorcnt(0);
      cur ^= 1;
    }
  }

  // Normalize, write bf16 att output in [B,T,C] (C index = h*64 + d)
#pragma unroll
  for (int r = 0; r < 8; ++r) {
    float inv = 1.0f / lrow[r];
    int qrow = qBase + hl * 8 + r;
    size_t base = ((size_t)bb * Tsz + qrow) * Csz + hh * HD;
#pragma unroll
    for (int dt = 0; dt < 4; ++dt)
      O[base + dt * 16 + l16] = f2bf(accO[dt][r] * inv);
  }
}

// ---------------------------------------------------------------------------
// Launcher
// ---------------------------------------------------------------------------
extern "C" void kernel_launch(void* const* d_in, const int* in_sizes, int n_in,
                              void* d_out, int out_size, void* d_ws, size_t ws_size,
                              hipStream_t stream) {
  const float* x     = (const float*)d_in[0];   // [B,T,C]
  const float* Wqkv  = (const float*)d_in[1];   // [C,3C]
  const float* bqkv  = (const float*)d_in[2];   // [3C]
  const float* Wproj = (const float*)d_in[3];   // [C,C]
  const float* bproj = (const float*)d_in[4];   // [C]
  float* out = (float*)d_out;

  const size_t nX    = (size_t)Bsz * Tsz * Csz;      // 4,194,304
  const size_t nWqkv = (size_t)Csz * 3 * Csz;        // 3,145,728
  const size_t nWprj = (size_t)Csz * Csz;            // 1,048,576

  char* ws = (char*)d_ws;
  size_t off = 0;
  auto take = [&](size_t elems) {
    u16* p = (u16*)(ws + off);
    off += ((elems * sizeof(u16)) + 255u) & ~(size_t)255u;
    return p;
  };
  u16* x_bf     = take(nX);
  u16* wqkv_bf  = take(nWqkv);
  u16* wproj_bf = take(nWprj);
  u16* q_bf     = take(nX);   // [B,H,T,HD]
  u16* k_bf     = take(nX);
  u16* v_bf     = take(nX);
  u16* att_bf   = take(nX);   // [B,T,C]

  // 1. Convert inputs to bf16
  cvt_bf16x4<<<(int)((nX / 4 + 255) / 256), 256, 0, stream>>>(
      (const float4*)x, (uint2*)x_bf, (int)(nX / 4));
  cvt_bf16x4<<<(int)((nWqkv / 4 + 255) / 256), 256, 0, stream>>>(
      (const float4*)Wqkv, (uint2*)wqkv_bf, (int)(nWqkv / 4));
  cvt_bf16x4<<<(int)((nWprj / 4 + 255) / 256), 256, 0, stream>>>(
      (const float4*)Wproj, (uint2*)wproj_bf, (int)(nWprj / 4));

  // 2. QKV projection: [4096,1024] x [1024,3072] -> q/k/v bf16
  gemm_bf16_kernel<0><<<dim3(3 * Csz / 128, Bsz * Tsz / 64), 128, 0, stream>>>(
      x_bf, wqkv_bf, bqkv, q_bf, k_bf, v_bf, nullptr,
      Bsz * Tsz, 3 * Csz, Csz);

  // 3. Causal flash attention -> att_bf [B,T,C]
  attn_kernel<<<Bsz * Hn * (Tsz / 64), 128, 0, stream>>>(q_bf, k_bf, v_bf, att_bf);

  // 4. Output projection: [4096,1024] x [1024,1024] + bias -> f32 d_out
  gemm_bf16_kernel<1><<<dim3(Csz / 128, Bsz * Tsz / 64), 128, 0, stream>>>(
      att_bf, wproj_bf, bproj, nullptr, nullptr, nullptr, out,
      Bsz * Tsz, Csz, Csz);
}
